// NacNet_84817014161892
// MI455X (gfx1250) — matrix operand order repaired and checked by reading, not compiled, not run on hardware
//
#include <hip/hip_runtime.h>

// ---------------------------------------------------------------------------
// GCN (12 layers) + 2 dense heads for MI455X (gfx1250, wave32).
// GEMMs: v_wmma_f32_16x16x32_bf16, fragment-contiguous packed B,
// 2 M-tiles x TPW N-tiles per wave (8 WMMA per K-step at TPW=4).
// Edge aggregation via coalesced f32 global atomics.
// ---------------------------------------------------------------------------

typedef __attribute__((ext_vector_type(16))) __bf16        bf16x16;
typedef __attribute__((ext_vector_type(8)))  float         f32x8;
typedef __attribute__((ext_vector_type(4)))  unsigned int  u32x4;

#define NNODES 60000
#define BN_EPS 1e-5f
#define SLOPE  0.01f

__device__ __forceinline__ unsigned short f2bf(float f) {
    unsigned int u = __float_as_uint(f);
    unsigned int r = u + 0x7FFFu + ((u >> 16) & 1u);   // round to nearest even
    return (unsigned short)(r >> 16);
}
__device__ __forceinline__ float bf2f(unsigned short h) {
    return __uint_as_float(((unsigned int)h) << 16);
}

// ----------------------------- tiny utility kernels ------------------------

__global__ void zero_f32_kernel(float* __restrict__ p, int n) {
    int t = blockIdx.x * blockDim.x + threadIdx.x;
    if (t < n) p[t] = 0.0f;
}

__global__ void deg_scatter_kernel(const int* __restrict__ col,
                                   float* __restrict__ deg, int E) {
    int e = blockIdx.x * blockDim.x + threadIdx.x;
    if (e < E) atomicAdd(&deg[col[e]], 1.0f);
}

__global__ void dinv_kernel(const float* __restrict__ deg,
                            float* __restrict__ dinv, int n) {
    int v = blockIdx.x * blockDim.x + threadIdx.x;
    if (v < n) dinv[v] = rsqrtf(deg[v] + 1.0f);        // +1 = self loop
}

// x [N,3] f32 -> act0 [N,32] bf16 (K padded with zeros)
__global__ void convert_x_kernel(const float* __restrict__ x,
                                 unsigned short* __restrict__ act, int n) {
    int t = blockIdx.x * blockDim.x + threadIdx.x;
    if (t >= n * 32) return;
    int c = t & 31, r = t >> 5;
    act[t] = (c < 3) ? f2bf(x[r * 3 + c]) : (unsigned short)0;
}

// Pack W [K, Nout] f32 into fragment-contiguous WMMA-B layout:
//   dst[(((tile*nKblk + kblk)*32 + lane)*8 + j]
//     = bf16(W[k][n]) | bf16(W[k+1][n])<<16
//   with k = 32*kblk + 8*(lane>>4) + 2*(j&3) + 16*(j>>2), n = 16*tile + (lane&15).
// Each lane of the GEMM reads its whole fragment as two b128 loads.
__global__ void pack_w_kernel(const float* __restrict__ W,
                              unsigned int* __restrict__ dst,
                              int K, int Nout, int nKblk) {
    int t = blockIdx.x * blockDim.x + threadIdx.x;
    int total = (Nout >> 4) * nKblk * 256;
    if (t >= total) return;
    int j    = t & 7;
    int lane = (t >> 3) & 31;
    int blk  = t >> 8;                 // tile*nKblk + kblk
    int kblk = blk % nKblk;
    int tile = blk / nKblk;
    int laneM = lane & 15, laneHi = lane >> 4;
    int k = (kblk << 5) + 8 * laneHi + 2 * (j & 3) + 16 * (j >> 2);
    int n = (tile << 4) + laneM;
    unsigned int lo = (k     < K) ? f2bf(W[(size_t)k       * Nout + n]) : 0u;
    unsigned int hi = (k + 1 < K) ? f2bf(W[(size_t)(k + 1) * Nout + n]) : 0u;
    dst[t] = lo | (hi << 16);
}

// ----------------------------- WMMA GEMM -----------------------------------
// One wave -> 2 M-tiles x TPW N-tiles (32 x 16*TPW output block).
// act : bf16 activations as K-pair dwords, row stride Kpad/2 dwords.
// Wp  : packed B fragments (see pack_w_kernel).
// mode 0: outF[r*Nout+c] = dinv[r] * (A@B)            (GCN layers)
// mode 1: outB[r*Nout+c] = bf16(lrelu((A@B)+bias[c]))  (lin1)

union AFrag { unsigned int u[8]; u32x4 q[2]; bf16x16 v; };

template <int TPW>
__global__ __launch_bounds__(256) void wmma_gemm_kernel(
    const unsigned int* __restrict__ act,
    const unsigned int* __restrict__ Wp,
    int Kpad, int Nout,
    const float* __restrict__ dinv,
    const float* __restrict__ bias,
    float* __restrict__ outF,
    unsigned short* __restrict__ outB,
    int mode)
{
    const int wave   = threadIdx.x >> 5;
    const int lane   = threadIdx.x & 31;
    const int nTiles = Nout >> 4;
    const int job    = blockIdx.y * 8 + wave;
    if (job * TPW >= nTiles) return;                   // wave-uniform exit
    const int tile0  = job * TPW;
    const int laneM  = lane & 15;
    const int laneHi = lane >> 4;                      // 0 | 1
    const int m0     = blockIdx.x << 5;                // 32 rows per block-x
    const int nKblk  = Kpad >> 5;
    const int kd     = Kpad >> 1;

    const unsigned int* arow0 = act + (size_t)(m0 + laneM)      * kd;
    const unsigned int* arow1 = act + (size_t)(m0 + 16 + laneM) * kd;
    // Per-tile B stream: contiguous 256-dword blocks per (tile, kblk)
    const unsigned int* bbase = Wp + ((size_t)tile0 * nKblk * 256) + lane * 8;

    f32x8 acc[2][TPW];
#pragma unroll
    for (int m = 0; m < 2; ++m)
#pragma unroll
        for (int t = 0; t < TPW; ++t) acc[m][t] = (f32x8){0,0,0,0,0,0,0,0};

    for (int q = 0; q < nKblk; ++q) {
        // A (16-bit 16x32): dword j<4 -> K = 8*laneHi + 2j ; j>=4 -> +16
        AFrag a0, a1, b[TPW];
        a0.q[0] = *(const u32x4*)(arow0 + (q << 4) + 4 * laneHi);
        a0.q[1] = *(const u32x4*)(arow0 + (q << 4) + 8 + 4 * laneHi);
        a1.q[0] = *(const u32x4*)(arow1 + (q << 4) + 4 * laneHi);
        a1.q[1] = *(const u32x4*)(arow1 + (q << 4) + 8 + 4 * laneHi);
#pragma unroll
        for (int t = 0; t < TPW; ++t) {
            const unsigned int* bp = bbase + ((size_t)t * nKblk + q) * 256;
            b[t].q[0] = *(const u32x4*)(bp);
            b[t].q[1] = *(const u32x4*)(bp + 4);
        }
        if (q + 1 < nKblk) {
            __builtin_prefetch(arow0 + (q << 4) + 16, 0, 3);
            __builtin_prefetch(arow1 + (q << 4) + 16, 0, 3);
        }
#pragma unroll
        for (int t = 0; t < TPW; ++t) {
            acc[0][t] = __builtin_amdgcn_wmma_f32_16x16x32_bf16(
                            false, a0.v, false, b[t].v, (short)0, acc[0][t],
                            false, false);
            acc[1][t] = __builtin_amdgcn_wmma_f32_16x16x32_bf16(
                            false, a1.v, false, b[t].v, (short)0, acc[1][t],
                            false, false);
        }
    }

    // C/D layout: VGPR r -> M = r + 8*laneHi, N = laneM
    if (mode == 0) {
#pragma unroll
        for (int m = 0; m < 2; ++m)
#pragma unroll
        for (int t = 0; t < TPW; ++t) {
            int n0 = (tile0 + t) << 4;
#pragma unroll
            for (int r = 0; r < 8; ++r) {
                int ro = m0 + 16 * m + r + 8 * laneHi;
                outF[(size_t)ro * Nout + n0 + laneM] = acc[m][t][r] * dinv[ro];
            }
        }
    } else {
#pragma unroll
        for (int m = 0; m < 2; ++m)
#pragma unroll
        for (int t = 0; t < TPW; ++t) {
            int n0 = (tile0 + t) << 4;
            float bb = bias[n0 + laneM];
#pragma unroll
            for (int r = 0; r < 8; ++r) {
                int ro = m0 + 16 * m + r + 8 * laneHi;
                float v = acc[m][t][r] + bb;
                v = (v > 0.f) ? v : v * SLOPE;
                outB[(size_t)ro * Nout + n0 + laneM] = f2bf(v);
            }
        }
    }
}

// ----------------------------- edge scatter --------------------------------
// agg[col[e]][f] += u[row[e]][f]; f = lane-contiguous -> coalesced.
__global__ void scatter_kernel(const int* __restrict__ row,
                               const int* __restrict__ col,
                               const float* __restrict__ u,
                               float* __restrict__ agg,
                               int shiftF, int E)
{
    int t = blockIdx.x * blockDim.x + threadIdx.x;
    int e = t >> shiftF;
    if (e >= E) return;
    int f = t & ((1 << shiftF) - 1);
    int r = row[e], c = col[e];
    atomicAdd(&agg[((size_t)c << shiftF) + f], u[((size_t)r << shiftF) + f]);
}

// ----------------------------- BatchNorm -----------------------------------
// z = dinv[r]*(agg + u) + bias  (self loop folded: u already has one dinv)
__global__ void bn_stats_kernel(const float* __restrict__ agg,
                                const float* __restrict__ u,
                                const float* __restrict__ dinv,
                                const float* __restrict__ bias,
                                int Nout, int rowsPerBlk, int nRows,
                                float* __restrict__ colsum,
                                float* __restrict__ colsumsq)
{
    int c = blockIdx.x * blockDim.x + threadIdx.x;
    if (c >= Nout) return;
    int r0 = blockIdx.y * rowsPerBlk;
    int r1 = r0 + rowsPerBlk; if (r1 > nRows) r1 = nRows;
    float bb = bias[c], s = 0.f, s2 = 0.f;
    for (int r = r0; r < r1; ++r) {
        size_t i = (size_t)r * Nout + c;
        float z = dinv[r] * (agg[i] + u[i]) + bb;
        s += z; s2 += z * z;
    }
    atomicAdd(&colsum[c], s);
    atomicAdd(&colsumsq[c], s2);
}

__global__ void bn_apply_kernel(const float* __restrict__ agg,
                                const float* __restrict__ u,
                                const float* __restrict__ dinv,
                                const float* __restrict__ bias,
                                const float* __restrict__ gamma,
                                const float* __restrict__ beta,
                                const float* __restrict__ colsum,
                                const float* __restrict__ colsumsq,
                                int shiftF, int nRows, float invN,
                                unsigned short* __restrict__ actOut)
{
    int t = blockIdx.x * blockDim.x + threadIdx.x;
    if (t >= (nRows << shiftF)) return;
    int c = t & ((1 << shiftF) - 1);
    int r = t >> shiftF;
    float z   = dinv[r] * (agg[t] + u[t]) + bias[c];
    float mu  = colsum[c] * invN;
    float var = colsumsq[c] * invN - mu * mu;
    float y   = gamma[c] * (z - mu) * rsqrtf(var + BN_EPS) + beta[c];
    y = (y > 0.f) ? y : y * SLOPE;
    actOut[t] = f2bf(y);
}

// ----------------------------- lin2 (16 -> 3) ------------------------------
__global__ void lin2_kernel(const unsigned short* __restrict__ act,
                            const float* __restrict__ W2,
                            const float* __restrict__ b2,
                            float* __restrict__ out, int nRows)
{
    int v = blockIdx.x * blockDim.x + threadIdx.x;
    if (v >= nRows) return;
    float a0 = b2[0], a1 = b2[1], a2 = b2[2];
#pragma unroll
    for (int k = 0; k < 16; ++k) {
        float a = bf2f(act[(size_t)v * 16 + k]);
        a0 += a * W2[k * 3 + 0];
        a1 += a * W2[k * 3 + 1];
        a2 += a * W2[k * 3 + 2];
    }
    out[(size_t)v * 3 + 0] = a0;
    out[(size_t)v * 3 + 1] = a1;
    out[(size_t)v * 3 + 2] = a2;
}

// ----------------------------- host orchestration --------------------------

static inline int ilog2i(int x) { int s = 0; while ((1 << s) < x) ++s; return s; }
static inline int cdiv(long long a, int b) { return (int)((a + b - 1) / b); }

static void launch_gemm(const unsigned short* act, const unsigned int* Wp,
                        int Kpad, int Nout, const float* dinv, const float* bias,
                        float* outF, unsigned short* outB, int mode,
                        hipStream_t stream)
{
    const int N = NNODES;
    int nTiles = Nout / 16;
    if (nTiles >= 4) {
        dim3 gg(N / 32, cdiv(nTiles / 4, 8));
        wmma_gemm_kernel<4><<<gg, 256, 0, stream>>>(
            (const unsigned int*)act, Wp, Kpad, Nout, dinv, bias, outF, outB, mode);
    } else if (nTiles == 2) {
        dim3 gg(N / 32, 1);
        wmma_gemm_kernel<2><<<gg, 256, 0, stream>>>(
            (const unsigned int*)act, Wp, Kpad, Nout, dinv, bias, outF, outB, mode);
    } else {
        dim3 gg(N / 32, 1);
        wmma_gemm_kernel<1><<<gg, 256, 0, stream>>>(
            (const unsigned int*)act, Wp, Kpad, Nout, dinv, bias, outF, outB, mode);
    }
}

extern "C" void kernel_launch(void* const* d_in, const int* in_sizes, int n_in,
                              void* d_out, int out_size, void* d_ws, size_t ws_size,
                              hipStream_t stream)
{
    (void)n_in; (void)out_size; (void)ws_size;
    const int H[15] = {3,32,64,128,256,256,512,512,256,256,128,64,32,16,3};
    const int N = NNODES;
    const int E = in_sizes[1] / 2;

    const float* x    = (const float*)d_in[0];
    const int*   erow = (const int*)d_in[1];
    const int*   ecol = erow + E;

    const float *W[13], *bias[13], *gamma[12], *beta[12];
    for (int i = 0; i < 12; ++i) {
        W[i]     = (const float*)d_in[2  + i];
        bias[i]  = (const float*)d_in[14 + i];
        gamma[i] = (const float*)d_in[26 + i];
        beta[i]  = (const float*)d_in[38 + i];
    }
    W[12]    = (const float*)d_in[50];   // lin1_W [32,16]
    bias[12] = (const float*)d_in[51];   // lin1_b
    const float* W2 = (const float*)d_in[52];
    const float* b2 = (const float*)d_in[53];

    // ---- workspace layout (bytes) ----
    char* ws = (char*)d_ws;
    float*          u        = (float*)(ws);                         // N*512 f32
    float*          agg      = (float*)(ws + 122880000);             // N*512 f32
    unsigned short* actA     = (unsigned short*)(ws + 245760000);    // N*512 bf16
    unsigned short* actB     = (unsigned short*)(ws + 307200000);    // N*512 bf16
    float*          deg      = (float*)(ws + 368640000);             // N f32
    float*          dinv     = (float*)(ws + 368880000);             // N f32
    float*          colsum   = (float*)(ws + 369120000);             // 512 f32
    float*          colsumsq = (float*)(ws + 369122048);             // 512 f32
    unsigned int*   Wp       = (unsigned int*)(ws + 369124096);      // packed weights

    // ---- pack weights (12 GCN + lin1) into fragment-contiguous B layout ----
    int wOff[13];
    {
        int off = 0;
        for (int l = 0; l < 13; ++l) {
            int K = H[l], Nout = H[l + 1];
            int Kpad  = ((K + 31) / 32) * 32;
            int nKblk = Kpad / 32;
            wOff[l] = off;
            int tot = (Nout / 16) * nKblk * 256;
            pack_w_kernel<<<cdiv(tot, 256), 256, 0, stream>>>(
                W[l], Wp + off, K, Nout, nKblk);
            off += tot;
        }
    }

    // ---- degree -> dinv ----
    zero_f32_kernel<<<cdiv(N, 256), 256, 0, stream>>>(deg, N);
    deg_scatter_kernel<<<cdiv(E, 256), 256, 0, stream>>>(ecol, deg, E);
    dinv_kernel<<<cdiv(N, 256), 256, 0, stream>>>(deg, dinv, N);

    // ---- x -> bf16 activations (K padded 3 -> 32) ----
    convert_x_kernel<<<cdiv((long long)N * 32, 256), 256, 0, stream>>>(x, actA, N);

    unsigned short* cur = actA;
    unsigned short* nxt = actB;
    const float invN = 1.0f / (float)N;

    for (int l = 0; l < 12; ++l) {
        int K = H[l], Nout = H[l + 1];
        int Kpad = ((K + 31) / 32) * 32;
        int sh   = ilog2i(Nout);

        // u = dinv ⊙ (act @ W)
        launch_gemm(cur, Wp + wOff[l], Kpad, Nout, dinv, nullptr, u, nullptr, 0, stream);

        zero_f32_kernel<<<cdiv((long long)N * Nout, 256), 256, 0, stream>>>(agg, N * Nout);
        zero_f32_kernel<<<4, 256, 0, stream>>>(colsum, 1024);   // colsum+colsumsq

        scatter_kernel<<<cdiv((long long)E << sh, 256), 256, 0, stream>>>(
            erow, ecol, u, agg, sh, E);

        int bsz = (Nout < 256) ? Nout : 256;
        dim3 gs(Nout / bsz, cdiv(N, 256));
        bn_stats_kernel<<<gs, bsz, 0, stream>>>(
            agg, u, dinv, bias[l], Nout, 256, N, colsum, colsumsq);

        bn_apply_kernel<<<cdiv((long long)N << sh, 256), 256, 0, stream>>>(
            agg, u, dinv, bias[l], gamma[l], beta[l],
            colsum, colsumsq, sh, N, invN, nxt);

        unsigned short* t = cur; cur = nxt; nxt = t;
    }

    // lin1: [N,32] @ [32,16] + b, lrelu -> bf16 (single N tile)
    launch_gemm(cur, Wp + wOff[12], 32, 16, nullptr, bias[12], nullptr, nxt, 1, stream);

    // lin2: [N,16] @ [16,3] + b -> d_out
    lin2_kernel<<<cdiv(N, 256), 256, 0, stream>>>(nxt, W2, b2, (float*)d_out, N);
}
